// MoRec_Model_51075751084105
// MI455X (gfx1250) — matrix operand order repaired and checked by reading, not compiled, not run on hardware
//
#include <hip/hip_runtime.h>
#include <hip/hip_bf16.h>

#define BB   32
#define SS   1024
#define DD   256
#define HH   4
#define DKH  64
#define LLAY 2
#define DF   1024
#define MMTOK (BB * SS)        // 32768 token rows
#define NEGF (-1e9f)

typedef _Float16 h16;
typedef __attribute__((ext_vector_type(16))) _Float16 v16h;
typedef __attribute__((ext_vector_type(8)))  _Float16 v8h;
typedef __attribute__((ext_vector_type(8)))  float    v8f;
typedef __attribute__((ext_vector_type(4)))  int      v4i;

// -------- optional gfx1250 async global->LDS path (guarded; safe fallback) ---
#if defined(__has_builtin)
#if __has_builtin(__builtin_amdgcn_global_load_async_to_lds_b128)
#define HAVE_ASYNC_LDS 1
#endif
#endif
#ifndef HAVE_ASYNC_LDS
#define HAVE_ASYNC_LDS 0
#endif

#if HAVE_ASYNC_LDS
__device__ __forceinline__ void async_cp16(const h16* g, h16* l) {
  __builtin_amdgcn_global_load_async_to_lds_b128(
      (__attribute__((address_space(1))) v4i*)g,
      (__attribute__((address_space(3))) v4i*)l, 0, 0);
}
__device__ __forceinline__ void wait_async0() {
#if __has_builtin(__builtin_amdgcn_s_wait_asynccnt)
  __builtin_amdgcn_s_wait_asynccnt(0);
#else
  asm volatile("s_wait_asynccnt 0" ::: "memory");
#endif
}
#endif

// ---------------- WMMA fragment helpers (wave32, 16x16x32 f16) ----------------
// A-matrix 16x32 f16 layout (ISA 7.12.2): lanes 0-15 row=lane, K {0..7,16..23};
// lanes 16-31 row=lane-16, K {8..15,24..31}. Caller passes p = row_base + koff,
// where koff = (lane>>4)*8; halves 0..7 at p, halves 8..15 at p+16.
__device__ __forceinline__ v16h load_fragA(const h16* p) {
  v8h lo = *(const v8h*)p;
  v8h hi = *(const v8h*)(p + 16);
  v16h f;
#pragma unroll
  for (int i = 0; i < 8; ++i) { f[i] = lo[i]; f[i + 8] = hi[i]; }
  return f;
}
// B-matrix 32x16 f16 (column-per-lane): lanes 0-15 hold K {0..7,16..23},
// lanes 16-31 hold K {8..15,24..31}. Caller passes p = col_base + (lane>>4)*8.
__device__ __forceinline__ v16h load_fragB(const h16* p) {
  v8h lo = *(const v8h*)p;
  v8h hi = *(const v8h*)(p + 16);
  v16h f;
#pragma unroll
  for (int i = 0; i < 8; ++i) { f[i] = lo[i]; f[i + 8] = hi[i]; }
  return f;
}
__device__ __forceinline__ v8f wmma32(v16h a, v16h b, v8f c) {
  return __builtin_amdgcn_wmma_f32_16x16x32_f16(false, a, false, b, (short)0, c,
                                                false, false);
}

// ---------------- shuffle reductions (wave32) ----------------
__device__ __forceinline__ float grp_max16(float v) {   // within 16-lane half
  v = fmaxf(v, __shfl_xor(v, 1, 32));
  v = fmaxf(v, __shfl_xor(v, 2, 32));
  v = fmaxf(v, __shfl_xor(v, 4, 32));
  v = fmaxf(v, __shfl_xor(v, 8, 32));
  return v;
}
__device__ __forceinline__ float grp_sum16(float v) {
  v += __shfl_xor(v, 1, 32); v += __shfl_xor(v, 2, 32);
  v += __shfl_xor(v, 4, 32); v += __shfl_xor(v, 8, 32);
  return v;
}
__device__ __forceinline__ float wave_sum32(float v) {
  v += __shfl_xor(v, 1, 32);  v += __shfl_xor(v, 2, 32);
  v += __shfl_xor(v, 4, 32);  v += __shfl_xor(v, 8, 32);
  v += __shfl_xor(v, 16, 32);
  return v;
}

// ---------------- weight convert + transpose: W[K,N] f32 -> Wt[N,K] f16 ------
__global__ void w_to_f16T(const float* __restrict__ W, h16* __restrict__ Wt,
                          int K, int N) {
  int idx = blockIdx.x * 256 + threadIdx.x;
  if (idx >= K * N) return;
  int k = idx / N, n = idx - k * N;
  Wt[(size_t)n * K + k] = (h16)W[idx];
}

// ---------------- register-blocked WMMA GEMM: Y = A[M,K] * Bt[N,K]^T ---------
// Block (4 waves) computes a 64x64 output tile; each wave 16 rows x 64 cols.
// One A fragment feeds 4 WMMAs. Weight tile (64 cols x 32 K = 4KB) is staged
// into double-buffered LDS with async global->LDS copies when available.
// MODE 0: f32 out        MODE 1: f32 out + bias
// MODE 2: relu(+bias) -> f16 out
// MODE 3: f16 out, head-major [B,H,S,dk]   (Q/K projections)
// MODE 4: f16 out, transposed [B,H,dk,S]   (V projection)
template <int MODE>
__global__ __launch_bounds__(128) void wmma_gemm(
    const h16* __restrict__ A, const h16* __restrict__ Bt,
    const float* __restrict__ bias, void* __restrict__ outp,
    int M, int N, int K) {
  int blocksN = N >> 6;
  int tM = blockIdx.x / blocksN;
  int tN = blockIdx.x - tM * blocksN;
  int w = threadIdx.x >> 5;
  int lane = threadIdx.x & 31;
  int grp = lane >> 4, li = lane & 15;
  const h16* ap = A + (size_t)(tM * 64 + w * 16 + li) * K + grp * 8;
  v8f c0 = {}, c1 = {}, c2 = {}, c3 = {};

#if HAVE_ASYNC_LDS
  __shared__ alignas(16) h16 lds_b[2][64][32];
  int lc  = threadIdx.x >> 1;          // column 0..63 handled by this thread
  int seg = (threadIdx.x & 1) * 16;    // first/second 16 halves of the K-step
  const h16* gB = Bt + (size_t)(tN * 64 + lc) * K + seg;
  async_cp16(gB + 0, &lds_b[0][lc][seg]);       // prologue: stage kk = 0
  async_cp16(gB + 8, &lds_b[0][lc][seg + 8]);
  int buf = 0;
  for (int kk = 0; kk < K; kk += 32) {
    wait_async0();                      // my copies for buf done
    __syncthreads();                    // whole tile staged; prev buf free
    if (kk + 32 < K) {                  // overlap next stage with compute
      async_cp16(gB + kk + 32,     &lds_b[buf ^ 1][lc][seg]);
      async_cp16(gB + kk + 32 + 8, &lds_b[buf ^ 1][lc][seg + 8]);
    }
    __builtin_prefetch(ap + kk + 256, 0, 1);
    v16h a = load_fragA(ap + kk);
    c0 = wmma32(a, load_fragB((const h16*)&lds_b[buf][ 0 + li][grp * 8]), c0);
    c1 = wmma32(a, load_fragB((const h16*)&lds_b[buf][16 + li][grp * 8]), c1);
    c2 = wmma32(a, load_fragB((const h16*)&lds_b[buf][32 + li][grp * 8]), c2);
    c3 = wmma32(a, load_fragB((const h16*)&lds_b[buf][48 + li][grp * 8]), c3);
    buf ^= 1;
  }
#else
  const h16* bp = Bt + (size_t)(tN * 64 + li) * K + grp * 8;
  for (int kk = 0; kk < K; kk += 32) {
    __builtin_prefetch(ap + kk + 256, 0, 1);
    v16h a = load_fragA(ap + kk);
    c0 = wmma32(a, load_fragB(bp + kk), c0);
    c1 = wmma32(a, load_fragB(bp + (size_t)16 * K + kk), c1);
    c2 = wmma32(a, load_fragB(bp + (size_t)32 * K + kk), c2);
    c3 = wmma32(a, load_fragB(bp + (size_t)48 * K + kk), c3);
  }
#endif

  v8f cc[4] = {c0, c1, c2, c3};
#pragma unroll
  for (int j = 0; j < 4; ++j) {
    int n = tN * 64 + j * 16 + li;
#pragma unroll
    for (int r = 0; r < 8; ++r) {
      int m = tM * 64 + w * 16 + r + grp * 8;   // C layout: VGPR r -> row r/r+8
      float val = cc[j][r];
      if (MODE == 0) {
        ((float*)outp)[(size_t)m * N + n] = val;
      } else if (MODE == 1) {
        ((float*)outp)[(size_t)m * N + n] = val + bias[n];
      } else if (MODE == 2) {
        float v = val + bias[n];
        ((h16*)outp)[(size_t)m * N + n] = (h16)(v > 0.f ? v : 0.f);
      } else if (MODE == 3) {
        int b_ = m >> 10, s_ = m & (SS - 1);
        int h_ = n >> 6,  j_ = n & (DKH - 1);
        ((h16*)outp)[(((size_t)b_ * HH + h_) * SS + s_) * DKH + j_] = (h16)val;
      } else {  // MODE 4
        int b_ = m >> 10, s_ = m & (SS - 1);
        int h_ = n >> 6,  j_ = n & (DKH - 1);
        ((h16*)outp)[(((size_t)b_ * HH + h_) * DKH + j_) * SS + s_] = (h16)val;
      }
    }
  }
}

// ---------------- fused flash attention (causal + padding mask) ---------------
// One wave owns a 16-row query tile of one (b,h); online softmax over 32-key
// blocks. P goes through LDS to re-shape C-layout -> A-fragment layout.
__global__ __launch_bounds__(128) void attn_kernel(
    const h16* __restrict__ Q, const h16* __restrict__ Kh,
    const h16* __restrict__ Vt, const int* __restrict__ mask,
    h16* __restrict__ Oh) {
  __shared__ alignas(16) h16 pbuf[4][16][32];
  int w = threadIdx.x >> 5;
  int wid = blockIdx.x * 4 + w;
  const int QT = SS / 16;
  int bh = wid / QT, qt = wid - bh * QT;
  int b = bh >> 2;                           // H = 4
  int lane = threadIdx.x & 31, grp = lane >> 4, li = lane & 15;
  int koff = grp * 8;
  int qbase = qt * 16;
  const h16* qp = Q + (size_t)bh * SS * DKH;
  const h16* kp = Kh + (size_t)bh * SS * DKH;
  const h16* vp = Vt + (size_t)bh * DKH * SS;
  const int* mp = mask + b * SS;

  v16h aq0 = load_fragA(qp + (size_t)(qbase + li) * DKH + koff);       // K 0..31
  v16h aq1 = load_fragA(qp + (size_t)(qbase + li) * DKH + 32 + koff);  // K 32..63

  v8f o0 = {}, o1 = {}, o2 = {}, o3 = {};
  float mrun[8], lrun[8];
#pragma unroll
  for (int r = 0; r < 8; ++r) { mrun[r] = -1e30f; lrun[r] = 0.f; }

  for (int kb = 0; kb < qbase + 16; kb += 32) {
    v8f s0 = {}, s1 = {};
    {
      const h16* kr = kp + (size_t)(kb + li) * DKH + grp * 8;
      s0 = wmma32(aq0, load_fragB(kr), s0);
      s0 = wmma32(aq1, load_fragB(kr + 32), s0);
    }
    {
      const h16* kr = kp + (size_t)(kb + 16 + li) * DKH + grp * 8;
      s1 = wmma32(aq0, load_fragB(kr), s1);
      s1 = wmma32(aq1, load_fragB(kr + 32), s1);
    }
    int key0 = kb + li, key1 = kb + 16 + li;
    bool m0 = mp[key0] != 0, m1 = mp[key1] != 0;
#pragma unroll
    for (int r = 0; r < 8; ++r) {
      int row = qbase + r + grp * 8;
      float x0 = (m0 && key0 <= row) ? s0[r] * 0.125f : NEGF;  // 1/sqrt(64)
      float x1 = (m1 && key1 <= row) ? s1[r] * 0.125f : NEGF;
      float tm = grp_max16(fmaxf(x0, x1));
      float mnew = fmaxf(mrun[r], tm);
      float alpha = __expf(mrun[r] - mnew);
      float p0 = __expf(x0 - mnew);
      float p1 = __expf(x1 - mnew);
      float rs = grp_sum16(p0 + p1);
      lrun[r] = lrun[r] * alpha + rs;
      mrun[r] = mnew;
      o0[r] *= alpha; o1[r] *= alpha; o2[r] *= alpha; o3[r] *= alpha;
      pbuf[w][r + grp * 8][li]      = (h16)p0;
      pbuf[w][r + grp * 8][16 + li] = (h16)p1;
    }
    asm volatile("s_wait_dscnt 0" ::: "memory");   // LDS stores visible to reads
    v16h pa = load_fragA(&pbuf[w][li][koff]);      // P as A-fragment [16x32]
    o0 = wmma32(pa, load_fragB(vp + (size_t)(0  + li) * SS + kb + grp * 8), o0);
    o1 = wmma32(pa, load_fragB(vp + (size_t)(16 + li) * SS + kb + grp * 8), o1);
    o2 = wmma32(pa, load_fragB(vp + (size_t)(32 + li) * SS + kb + grp * 8), o2);
    o3 = wmma32(pa, load_fragB(vp + (size_t)(48 + li) * SS + kb + grp * 8), o3);
  }
  int h = bh & 3;
#pragma unroll
  for (int r = 0; r < 8; ++r) {
    float inv = 1.f / lrun[r];
    int row = qbase + r + grp * 8;
    size_t base = ((size_t)b * SS + row) * DD + h * DKH;
    Oh[base + 0  + li] = (h16)(o0[r] * inv);
    Oh[base + 16 + li] = (h16)(o1[r] * inv);
    Oh[base + 32 + li] = (h16)(o2[r] * inv);
    Oh[base + 48 + li] = (h16)(o3[r] * inv);
  }
}

// ---------------- residual add + LayerNorm (one wave per 256-wide row) --------
__global__ __launch_bounds__(256) void fused_add_ln(
    const float* __restrict__ xin, const float* __restrict__ addv,
    const float* __restrict__ g, const float* __restrict__ beta,
    float* __restrict__ outf, h16* __restrict__ outh, int pos_broadcast) {
  int row = blockIdx.x * blockDim.y + threadIdx.y;
  int lane = threadIdx.x;
  const float* xr = xin + (size_t)row * DD;
  const float* ar = addv + (pos_broadcast ? (size_t)(row & (SS - 1)) * DD
                                          : (size_t)row * DD);
  float v[8];
  float s = 0.f;
#pragma unroll
  for (int j = 0; j < 8; ++j) {
    int c = lane + 32 * j;
    v[j] = xr[c] + ar[c];
    s += v[j];
  }
  s = wave_sum32(s);
  float mean = s * (1.f / DD);
  float vs = 0.f;
#pragma unroll
  for (int j = 0; j < 8; ++j) { float d = v[j] - mean; vs += d * d; }
  vs = wave_sum32(vs);
  float inv = rsqrtf(vs * (1.f / DD) + 1e-6f);
#pragma unroll
  for (int j = 0; j < 8; ++j) {
    int c = lane + 32 * j;
    float o = (v[j] - mean) * inv * g[c] + beta[c];
    outf[(size_t)row * DD + c] = o;
    if (outh) outh[(size_t)row * DD + c] = (h16)o;
  }
}

// ---------------- host orchestration ----------------
extern "C" void kernel_launch(void* const* d_in, const int* in_sizes, int n_in,
                              void* d_out, int out_size, void* d_ws,
                              size_t ws_size, hipStream_t stream) {
  (void)in_sizes; (void)n_in; (void)out_size; (void)ws_size;
  const float* input_embs = (const float*)d_in[0];
  const int*   log_mask   = (const int*)d_in[1];
  const float* pos_emb    = (const float*)d_in[2];
  const float* ln0_g = (const float*)d_in[3];
  const float* ln0_b = (const float*)d_in[4];
  const float* Wq = (const float*)d_in[5];
  const float* Wk = (const float*)d_in[6];
  const float* Wv = (const float*)d_in[7];
  const float* Wo = (const float*)d_in[8];
  const float* ln1_g = (const float*)d_in[9];
  const float* ln1_b = (const float*)d_in[10];
  const float* W1 = (const float*)d_in[11];
  const float* b1 = (const float*)d_in[12];
  const float* W2 = (const float*)d_in[13];
  const float* b2 = (const float*)d_in[14];
  const float* ln2_g = (const float*)d_in[15];
  const float* ln2_b = (const float*)d_in[16];

  char* ws = (char*)d_ws;
  size_t off = 0;
  auto alloc = [&](size_t bytes) -> void* {
    void* p = ws + off;
    off += (bytes + 255) & ~(size_t)255;
    return p;
  };
  float* xf   = (float*)alloc((size_t)MMTOK * DD * 4);
  h16*   xh   = (h16*)  alloc((size_t)MMTOK * DD * 2);
  h16*   qh   = (h16*)  alloc((size_t)MMTOK * DD * 2);   // [B,H,S,dk]
  h16*   kh   = (h16*)  alloc((size_t)MMTOK * DD * 2);   // [B,H,S,dk]
  h16*   vth  = (h16*)  alloc((size_t)MMTOK * DD * 2);   // [B,H,dk,S]
  h16*   ohb  = (h16*)  alloc((size_t)MMTOK * DD * 2);   // [B,S,D]
  float* proj = (float*)alloc((size_t)MMTOK * DD * 4);
  h16*   h1b  = (h16*)  alloc((size_t)MMTOK * DF * 2);
  h16*   wqT  = (h16*)  alloc((size_t)LLAY * DD * DD * 2);
  h16*   wkT  = (h16*)  alloc((size_t)LLAY * DD * DD * 2);
  h16*   wvT  = (h16*)  alloc((size_t)LLAY * DD * DD * 2);
  h16*   woT  = (h16*)  alloc((size_t)LLAY * DD * DD * 2);
  h16*   w1T  = (h16*)  alloc((size_t)LLAY * DD * DF * 2);
  h16*   w2T  = (h16*)  alloc((size_t)LLAY * DF * DD * 2);

  // weight convert+transpose (all layers up front)
  for (int l = 0; l < LLAY; ++l) {
    size_t wo = (size_t)l * DD * DD;
    size_t wf = (size_t)l * DD * DF;
    int gDD = (DD * DD + 255) / 256;   // 256
    int gDF = (DD * DF + 255) / 256;   // 1024
    w_to_f16T<<<gDD, 256, 0, stream>>>(Wq + wo, wqT + wo, DD, DD);
    w_to_f16T<<<gDD, 256, 0, stream>>>(Wk + wo, wkT + wo, DD, DD);
    w_to_f16T<<<gDD, 256, 0, stream>>>(Wv + wo, wvT + wo, DD, DD);
    w_to_f16T<<<gDD, 256, 0, stream>>>(Wo + wo, woT + wo, DD, DD);
    w_to_f16T<<<gDF, 256, 0, stream>>>(W1 + wf, w1T + wf, DD, DF);
    w_to_f16T<<<gDF, 256, 0, stream>>>(W2 + wf, w2T + wf, DF, DD);
  }

  dim3 lnBlk(32, 8);
  int lnGrid = MMTOK / 8;                       // 4096
  // x = LN(input + pos)
  fused_add_ln<<<lnGrid, lnBlk, 0, stream>>>(input_embs, pos_emb, ln0_g, ln0_b,
                                             xf, xh, 1);

  const int gemmDD = (MMTOK / 64) * (DD / 64);  // 2048 blocks (64x64 tiles)
  const int gemmDF = (MMTOK / 64) * (DF / 64);  // 8192 blocks
  const int attnG  = BB * HH * (SS / 16) / 4;   // 2048 blocks

  for (int l = 0; l < LLAY; ++l) {
    size_t wo = (size_t)l * DD * DD;
    size_t wf = (size_t)l * DD * DF;
    // projections
    wmma_gemm<3><<<gemmDD, 128, 0, stream>>>(xh, wqT + wo, nullptr, qh,
                                             MMTOK, DD, DD);
    wmma_gemm<3><<<gemmDD, 128, 0, stream>>>(xh, wkT + wo, nullptr, kh,
                                             MMTOK, DD, DD);
    wmma_gemm<4><<<gemmDD, 128, 0, stream>>>(xh, wvT + wo, nullptr, vth,
                                             MMTOK, DD, DD);
    // fused attention
    attn_kernel<<<attnG, 128, 0, stream>>>(qh, kh, vth, log_mask, ohb);
    // output projection
    wmma_gemm<0><<<gemmDD, 128, 0, stream>>>(ohb, woT + wo, nullptr, proj,
                                             MMTOK, DD, DD);
    // x = LN(x + attn)
    fused_add_ln<<<lnGrid, lnBlk, 0, stream>>>(xf, proj, ln1_g + l * DD,
                                               ln1_b + l * DD, xf, xh, 0);
    // FFN
    wmma_gemm<2><<<gemmDF, 128, 0, stream>>>(xh, w1T + wf, b1 + l * DF, h1b,
                                             MMTOK, DF, DD);
    wmma_gemm<1><<<gemmDD, 128, 0, stream>>>(h1b, w2T + wf, b2 + l * DD, proj,
                                             MMTOK, DD, DF);
    // x = LN(x + ffn); final layer writes straight to d_out (f32)
    float* dstF = (l == LLAY - 1) ? (float*)d_out : xf;
    h16*   dstH = (l == LLAY - 1) ? nullptr : xh;
    fused_add_ln<<<lnGrid, lnBlk, 0, stream>>>(xf, proj, ln2_g + l * DD,
                                               ln2_b + l * DD, dstF, dstH, 0);
  }
}
